// WeightedFeatureInteraction_83769042141576
// MI455X (gfx1250) — matrix operand order repaired and checked by reading, not compiled, not run on hardware
//
#include <hip/hip_runtime.h>
#include <hip/hip_bf16.h>
#include <stdint.h>

// ---------------------------------------------------------------------------
// WeightedFeatureInteraction: out[b] = sum_{i<j} W[i,j] * <x[b,i,:], x[b,j,:]>
//   x: [8192, 64, 128] f32, W: [64,64] f32, out: [8192,1] f32
// Strategy: per-wave Gram matrix via v_wmma_f32_16x16x32_bf16 on upper-tri tiles.
//   - 256 MB single-pass stream -> non-temporal loads (don't churn 192MB L2)
//   - W (16 KB) stays L2-hot with regular temporal policy
//   - masked-weight apply is branch-free (unconditional load + select)
// ---------------------------------------------------------------------------

typedef __attribute__((ext_vector_type(16))) __bf16        v16bf;
typedef __attribute__((ext_vector_type(8)))  float         v8f;
typedef __attribute__((ext_vector_type(8)))  unsigned int  v8u;
typedef __attribute__((ext_vector_type(4)))  unsigned int  u32x4;

#define NUM_FIELDS 64
#define DIM        128
#define BATCH      8192
#define WAVES_PER_BLOCK 4

// Per-wave LDS tile: 64 rows x 128 bf16 = 64 rows x 256 B = 16 KB (4096 u32)
#define ROW_U32      64
#define WAVE_LDS_U32 (64 * ROW_U32)

// XOR swizzle at 32-byte granularity: row stride (256 B) equals the 64-bank
// period, so without swizzling all rows land on the same banks. Flipping byte
// bits [7:5] by (row & 7) spreads 8 consecutive rows across all 64 banks while
// preserving 8/16/32-byte access alignment (XOR value is a multiple of 32).
__device__ __forceinline__ unsigned swz_word(unsigned row, unsigned kbyte) {
    return row * ROW_U32 + ((kbyte ^ ((row & 7u) << 5)) >> 2);
}

// f32 -> bf16 round-to-nearest-even on raw bit patterns, packed pair -> u32.
__device__ __forceinline__ unsigned pack_bf16_bits(unsigned ua, unsigned ub) {
    ua += 0x7FFFu + ((ua >> 16) & 1u);
    ub += 0x7FFFu + ((ub >> 16) & 1u);
    return (ua >> 16) | (ub & 0xFFFF0000u);
}

// A fragment (16x32 bf16, tile row-block t, K-chunk kc).
// ISA layout (CDNA5 ISA, 16-bit A 16x32): lane half h = lane/16,
// row M = lane%16; VGPRs 0-3 hold K = 8h+0..7, VGPRs 4-7 hold K = 16+8h+0..7.
// => per lane: two contiguous 16-byte chunks from row (t*16 + M).
__device__ __forceinline__ v16bf load_fragA(const unsigned* my, unsigned t,
                                            unsigned kc, unsigned lane) {
    const unsigned m   = lane & 15u;
    const unsigned h   = lane >> 4;
    const unsigned row = t * 16u + m;
    const unsigned k0  = kc * 64u + h * 16u;          // byte offset of K-chunk
    const uint4 lo = *reinterpret_cast<const uint4*>(my + swz_word(row, k0));
    const uint4 hi = *reinterpret_cast<const uint4*>(my + swz_word(row, k0 + 32u));
    v8u u;
    u[0] = lo.x; u[1] = lo.y; u[2] = lo.z; u[3] = lo.w;
    u[4] = hi.x; u[5] = hi.y; u[6] = hi.z; u[7] = hi.w;
    return __builtin_bit_cast(v16bf, u);
}

// B fragment (32x16 bf16 = x^T tile): column N of B is row (t*16 + N) of x.
// Row-striped-across-lanes layout: lane holds col N = lane%16; VGPRs 0-3 hold
// K = 8h+0..7, VGPRs 4-7 hold K = 16+8h+0..7 (same as A) => two contiguous
// 16-byte chunks per lane from row (t*16 + N). No transpose ever materialized.
__device__ __forceinline__ v16bf load_fragB(const unsigned* my, unsigned t,
                                            unsigned kc, unsigned lane) {
    const unsigned n   = lane & 15u;
    const unsigned h   = lane >> 4;
    const unsigned row = t * 16u + n;
    const unsigned k0  = kc * 64u + h * 16u;
    const uint4 lo = *reinterpret_cast<const uint4*>(my + swz_word(row, k0));
    const uint4 hi = *reinterpret_cast<const uint4*>(my + swz_word(row, k0 + 32u));
    v8u u;
    u[0] = lo.x; u[1] = lo.y; u[2] = lo.z; u[3] = lo.w;
    u[4] = hi.x; u[5] = hi.y; u[6] = hi.z; u[7] = hi.w;
    return __builtin_bit_cast(v16bf, u);
}

__global__ __launch_bounds__(WAVES_PER_BLOCK * 32, 1)
void wfi_gram_wmma_kernel(const float* __restrict__ x,
                          const float* __restrict__ W,
                          float* __restrict__ out) {
    __shared__ unsigned lds[WAVES_PER_BLOCK * WAVE_LDS_U32];   // 64 KB

    const unsigned tid  = threadIdx.x;
    const unsigned lane = tid & 31u;
    const unsigned wave = tid >> 5;
    const unsigned b    = blockIdx.x * WAVES_PER_BLOCK + wave;

    unsigned* my = lds + wave * WAVE_LDS_U32;
    const float* __restrict__ xb = x + (size_t)b * (NUM_FIELDS * DIM);

    // ---- Stage: global f32 -> bf16 -> LDS (swizzled). One row per iteration,
    // lane L covers elements 4L..4L+3 of the row (fully coalesced b128 loads).
    // Non-temporal: single-pass 256MB stream must not evict the hot W matrix.
    #pragma unroll 4
    for (unsigned it = 0; it < 64u; ++it) {
        if (it + 8u < 64u)   // speculative prefetch ahead (global_prefetch_b8)
            __builtin_prefetch(xb + (it + 8u) * DIM + lane * 4u, 0, 0);
        const u32x4 v = __builtin_nontemporal_load(
            reinterpret_cast<const u32x4*>(xb + it * DIM) + lane);
        uint2 p;
        p.x = pack_bf16_bits(v.x, v.y);
        p.y = pack_bf16_bits(v.z, v.w);
        // byte offset lane*8 within row 'it'; swizzle keeps 8B alignment.
        *reinterpret_cast<uint2*>(my + swz_word(it, lane * 8u)) = p;
    }
    // Producer and consumer are the same wave: DScnt ordering is inserted by
    // the compiler; no cross-wave sharing, so no workgroup barrier needed.

    // ---- Gram tiles, upper triangle only (strict-upper weight mask kills the
    // rest, and the i<j mask inside diagonal tiles is applied per element).
    const unsigned TI[10] = {0,0,0,0,1,1,1,2,2,3};
    const unsigned TJ[10] = {0,1,2,3,1,2,3,2,3,3};

    const unsigned h    = lane >> 4;
    const unsigned nloc = lane & 15u;
    float acc = 0.0f;

    #pragma unroll
    for (int p = 0; p < 10; ++p) {
        const unsigned ti = TI[p];
        const unsigned tj = TJ[p];

        v8f c = {};
        #pragma unroll
        for (unsigned kc = 0; kc < 4u; ++kc) {
            const v16bf a  = load_fragA(my, ti, kc, lane);
            const v16bf bb = load_fragB(my, tj, kc, lane);
            c = __builtin_amdgcn_wmma_f32_16x16x32_bf16(
                    /*neg_a=*/false, a, /*neg_b=*/false, bb,
                    /*c_mod=*/(short)0, c, /*reuse_a=*/false, /*reuse_b=*/false);
        }

        // Accumulator layout: lane half h, VGPR r -> G element
        //   (M = r + 8h within tile ti, N = lane%16 within tile tj).
        // Branch-free: i,j < 64 always, so load unconditionally (L2-hot W)
        // and mask with a select -> v_cndmask, no exec-mask branching.
        const unsigned j = tj * 16u + nloc;
        #pragma unroll
        for (int r = 0; r < 8; ++r) {
            const unsigned i  = ti * 16u + (unsigned)r + 8u * h;
            const float    wv = W[i * NUM_FIELDS + j];
            const float    w  = (i < j) ? wv : 0.0f;
            acc = fmaf(c[r], w, acc);
        }
    }

    // ---- 32-lane reduction (wave32).
    #pragma unroll
    for (int off = 16; off > 0; off >>= 1)
        acc += __shfl_down(acc, off, 32);

    if (lane == 0)
        __builtin_nontemporal_store(acc, out + b);
}

extern "C" void kernel_launch(void* const* d_in, const int* in_sizes, int n_in,
                              void* d_out, int out_size, void* d_ws, size_t ws_size,
                              hipStream_t stream) {
    (void)in_sizes; (void)n_in; (void)d_ws; (void)ws_size; (void)out_size;
    const float* x   = (const float*)d_in[0];   // [8192, 8192] f32
    const float* W   = (const float*)d_in[1];   // [64, 64] f32
    float*       out = (float*)d_out;           // [8192, 1] f32

    dim3 grid(BATCH / WAVES_PER_BLOCK);         // 2048 blocks
    dim3 block(WAVES_PER_BLOCK * 32);           // 4 waves (wave32)
    hipLaunchKernelGGL(wfi_gram_wmma_kernel, grid, block, 0, stream, x, W, out);
}
